// GraniteMultiHeadLatentAttention_45105746543205
// MI455X (gfx1250) — matrix operand order repaired and checked by reading, not compiled, not run on hardware
//
#include <hip/hip_runtime.h>

// ---------------------------------------------------------------------------
// MLA forward for MI455X (gfx1250, wave32, WMMA + Tensor Data Mover).
// Pipeline: f32->bf16 convert (weights transposed once), double-buffered
// TDM-staged WMMA GEMMs, double-buffered TDM-staged causal flash attention.
// ---------------------------------------------------------------------------

#define HIDDEN_  2048
#define SEQ_     2048
#define BATCH_   2
#define NHEADS_  32
#define HEADDIM_ 64
#define QCOMP_   1024
#define KVCOMP_  512
#define ROWS_    (BATCH_ * SEQ_)   // 4096

typedef __attribute__((ext_vector_type(16))) __bf16 v16bf;
typedef __attribute__((ext_vector_type(8)))  float  v8f;
typedef __attribute__((ext_vector_type(4)))  unsigned int u32x4;
typedef __attribute__((ext_vector_type(4)))  int i32x4;
typedef __attribute__((ext_vector_type(8)))  int i32x8;

union AFrag { unsigned int u[8]; v16bf v; };
union CFrag { float f[8];        v8f  v; };

#if __has_builtin(__builtin_amdgcn_tensor_load_to_lds) && __has_builtin(__builtin_amdgcn_s_wait_tensorcnt)
#define USE_TDM 1
#else
#define USE_TDM 0
#endif

__device__ __forceinline__ unsigned short f32_to_bf16(float x) {
  unsigned int u = __float_as_uint(x);
  u += 0x7FFFu + ((u >> 16) & 1u);   // round-to-nearest-even
  return (unsigned short)(u >> 16);
}

__device__ __forceinline__ float rowmax16(float x) {
  x = fmaxf(x, __shfl_xor(x, 1, 32));
  x = fmaxf(x, __shfl_xor(x, 2, 32));
  x = fmaxf(x, __shfl_xor(x, 4, 32));
  x = fmaxf(x, __shfl_xor(x, 8, 32));
  return x;
}
__device__ __forceinline__ float rowsum16(float x) {
  x += __shfl_xor(x, 1, 32);
  x += __shfl_xor(x, 2, 32);
  x += __shfl_xor(x, 4, 32);
  x += __shfl_xor(x, 8, 32);
  return x;
}

#if USE_TDM
// 2D bf16 tile DMA: global (tile_h rows x tile_w elems, row stride `stride`
// elems) -> LDS packed row-major. D# per CDNA5 ISA.
__device__ __forceinline__ void tdm_load_tile_bf16(unsigned short* lds_dst,
                                                   const unsigned short* gsrc,
                                                   unsigned tile_w, unsigned tile_h,
                                                   unsigned stride)
{
  const unsigned lds_addr = (unsigned)(unsigned long)(void*)lds_dst;   // addr[31:0] = LDS offset
  const unsigned long ga  = (unsigned long)(const void*)gsrc;
  u32x4 g0;
  g0.x = 0x1u;                                                  // count=1, user-mode
  g0.y = lds_addr;                                              // lds_addr
  g0.z = (unsigned)ga;                                          // global_addr[31:0]
  g0.w = (unsigned)((ga >> 32) & 0x01FFFFFFu) | 0x80000000u;    // global_addr[56:32] | type=2
  i32x8 g1;
  g1[0] = (int)(1u << 16);                                      // data_size=1 (2B); mask=0
  g1[1] = (int)((tile_w & 0xFFFFu) << 16);                      // tensor_dim0[15:0]
  g1[2] = (int)((tile_w >> 16) | ((tile_h & 0xFFFFu) << 16));   // td0 hi | td1 lo
  g1[3] = (int)((tile_h >> 16) | (tile_w << 16));               // td1 hi | tile_dim0
  g1[4] = (int)(tile_h & 0xFFFFu);                              // tile_dim1 | tile_dim2=0
  g1[5] = (int)stride;                                          // dim0_stride[31:0]
  g1[6] = 0;                                                    // dim0_stride hi | dim1_stride lo
  g1[7] = 0;
  const i32x4 z4 = {0, 0, 0, 0};
#if __clang_major__ >= 23
  const i32x8 z8 = {0, 0, 0, 0, 0, 0, 0, 0};
  __builtin_amdgcn_tensor_load_to_lds(g0, g1, z4, z4, z8, 0);
#else
  __builtin_amdgcn_tensor_load_to_lds(g0, g1, z4, z4, 0);
#endif
}
#endif

// ---------------------------------------------------------------------------
__global__ __launch_bounds__(256)
void cvt_f32_to_bf16(const float* __restrict__ in, unsigned short* __restrict__ out, int n) {
  const int idx = (blockIdx.x * 256 + threadIdx.x) * 4;
  if (idx + 3 < n) {
    const float4 v = *(const float4*)(in + idx);
    ushort4 o;
    o.x = f32_to_bf16(v.x); o.y = f32_to_bf16(v.y);
    o.z = f32_to_bf16(v.z); o.w = f32_to_bf16(v.w);
    *(ushort4*)(out + idx) = o;
  }
}

// in[K][N] f32 -> out[N][K] bf16 (one-time weight transpose, LDS-tiled)
__global__ __launch_bounds__(256)
void cvt_transpose_f32_to_bf16(const float* __restrict__ in, unsigned short* __restrict__ out,
                               int K, int N) {
  __shared__ unsigned short t[32][33];
  const int bk = blockIdx.y * 32, bn = blockIdx.x * 32;
  const int tx = threadIdx.x & 31, ty = threadIdx.x >> 5;   // ty: 0..7
#pragma unroll
  for (int i = 0; i < 32; i += 8)
    t[ty + i][tx] = f32_to_bf16(in[(long)(bk + ty + i) * N + bn + tx]);
  __syncthreads();
#pragma unroll
  for (int i = 0; i < 32; i += 8)
    out[(long)(bn + ty + i) * K + bk + tx] = t[tx][ty + i];
}

// ---------------------------------------------------------------------------
// C[M,N] = A[M,K] (bf16 row-major, stride lda) x Bt[N,K]^T (bf16 row-major).
// Block tile 128x128x32, 8 waves, wave tile 32x64. Double-buffered LDS,
// TDM-staged (wave 0 issues DMA, waits TENSORcnt before the barrier).
// OUT_MODE: 0 = f32, 1 = bf16, 2 = bf16 transposed per batch ([b][n][s]).
// ---------------------------------------------------------------------------
template <int OUT_MODE>
__global__ __launch_bounds__(256, 2)
void gemm_bf16_wmma(const unsigned short* __restrict__ A,
                    const unsigned short* __restrict__ Bt,
                    void* __restrict__ Cp,
                    int K, int lda, int ldbt, int ldc)
{
  __shared__ unsigned short As[2][128][32];   // [buf][m][k]
  __shared__ unsigned short Bs[2][128][32];   // [buf][n][k]

  const int tid  = threadIdx.x;
  const int lane = tid & 31;
  const int wid  = tid >> 5;
  const int wm   = (wid & 3) * 32;
  const int wn   = (wid >> 2) * 64;
  const int l15  = lane & 15;
  const int lhi  = lane >> 4;
  const long tile_m = (long)blockIdx.y * 128;
  const long tile_n = (long)blockIdx.x * 128;

  CFrag acc[2][4];
#pragma unroll
  for (int mt = 0; mt < 2; ++mt)
#pragma unroll
    for (int nt = 0; nt < 4; ++nt)
#pragma unroll
      for (int r = 0; r < 8; ++r) acc[mt][nt].f[r] = 0.0f;

  auto stage = [&](int k0, int bufi) {
#if USE_TDM
    if (wid == 0) {
      tdm_load_tile_bf16(&As[bufi][0][0], A  + tile_m * (long)lda  + k0, 32u, 128u, (unsigned)lda);
      tdm_load_tile_bf16(&Bs[bufi][0][0], Bt + tile_n * (long)ldbt + k0, 32u, 128u, (unsigned)ldbt);
    }
#else
    const int r = tid >> 1, c = (tid & 1) * 16;
    {
      const unsigned short* src = A + (tile_m + r) * (long)lda + k0 + c;
      *(uint4*)&As[bufi][r][c]     = *(const uint4*)src;
      *(uint4*)&As[bufi][r][c + 8] = *(const uint4*)(src + 8);
    }
    {
      const unsigned short* src = Bt + (tile_n + r) * (long)ldbt + k0 + c;
      *(uint4*)&Bs[bufi][r][c]     = *(const uint4*)src;
      *(uint4*)&Bs[bufi][r][c + 8] = *(const uint4*)(src + 8);
    }
#endif
  };

  stage(0, 0);
#if USE_TDM
  if (wid == 0) __builtin_amdgcn_s_wait_tensorcnt(0);
#endif
  __syncthreads();

  const int nk = K >> 5;
  for (int kb = 0; kb < nk; ++kb) {
    const int cur = kb & 1, nxt = cur ^ 1;
    if (kb + 1 < nk) stage((kb + 1) << 5, nxt);   // overlap DMA with WMMA

    AFrag a[2], b[4];
#pragma unroll
    for (int mt = 0; mt < 2; ++mt) {
      const int m = wm + mt * 16 + l15;
#pragma unroll
      for (int i = 0; i < 8; ++i) {
        const int k = ((i < 4) ? (2 * i) : (8 + 2 * i)) + 8 * lhi;   // A-frag k map
        a[mt].u[i] = *(const unsigned int*)&As[cur][m][k];
      }
    }
#pragma unroll
    for (int nt = 0; nt < 4; ++nt) {
      const int n = wn + nt * 16 + l15;
#pragma unroll
      for (int i = 0; i < 8; ++i) {
        const int k = ((i < 4) ? (2 * i) : (8 + 2 * i)) + 8 * lhi;   // B-frag k map
        b[nt].u[i] = *(const unsigned int*)&Bs[cur][n][k];
      }
    }
#pragma unroll
    for (int mt = 0; mt < 2; ++mt)
#pragma unroll
      for (int nt = 0; nt < 4; ++nt)
        acc[mt][nt].v = __builtin_amdgcn_wmma_f32_16x16x32_bf16(
            false, a[mt].v, false, b[nt].v, (short)0, acc[mt][nt].v, false, false);

#if USE_TDM
    if (wid == 0) __builtin_amdgcn_s_wait_tensorcnt(0);
#endif
    __syncthreads();
  }

  if (OUT_MODE == 2) {
    // vt[b][n][s]: per lane, r=0..7 are 8 consecutive s -> one b128 store
    unsigned short* vt = (unsigned short*)Cp;
#pragma unroll
    for (int mt = 0; mt < 2; ++mt)
#pragma unroll
      for (int nt = 0; nt < 4; ++nt) {
        const long n  = tile_n + wn + nt * 16 + l15;
        const long m0 = tile_m + wm + mt * 16 + 8 * lhi;
        const long bb = m0 >> 11, s = m0 & 2047;
        unsigned short pk[8];
#pragma unroll
        for (int r = 0; r < 8; ++r) pk[r] = f32_to_bf16(acc[mt][nt].f[r]);
        *(uint4*)(vt + bb * (long)(HIDDEN_ * SEQ_) + n * SEQ_ + s) = *(uint4*)&pk[0];
      }
  } else {
#pragma unroll
    for (int mt = 0; mt < 2; ++mt)
#pragma unroll
      for (int nt = 0; nt < 4; ++nt) {
        const long n = tile_n + wn + nt * 16 + l15;
#pragma unroll
        for (int r = 0; r < 8; ++r) {
          const long m = tile_m + wm + mt * 16 + r + 8 * lhi;
          const float v = acc[mt][nt].f[r];
          if (OUT_MODE == 1) ((unsigned short*)Cp)[m * ldc + n] = f32_to_bf16(v);
          else               ((float*)Cp)[m * ldc + n] = v;
        }
      }
  }
}

// ---------------------------------------------------------------------------
// Causal flash attention, HEAD_DIM = 64. Block = 4 waves x 16 q-rows.
// K staged row-major [key][d]; V staged from pre-transposed vt as [d][key].
// Double-buffered, TDM-staged.
// ---------------------------------------------------------------------------
__global__ __launch_bounds__(128, 4)
void mla_flash_attn(const unsigned short* __restrict__ Q,
                    const unsigned short* __restrict__ Kk,
                    const unsigned short* __restrict__ Vt,   // [b][h*64+d][s]
                    unsigned short* __restrict__ O)
{
  __shared__ unsigned short Ks[2][32][64];     // [buf][key][d]
  __shared__ unsigned short Vs[2][64][32];     // [buf][d][key]
  __shared__ unsigned short Ps[4][16][32];     // per-wave P re-layout buffer

  const int tid  = threadIdx.x;
  const int lane = tid & 31;
  const int w    = tid >> 5;
  const int l15  = lane & 15;
  const int lhi  = lane >> 4;
  const int q_base = blockIdx.x * 64;
  const int b = blockIdx.y >> 5;
  const int h = blockIdx.y & 31;
  const long seq_off = (long)b * SEQ_ * HIDDEN_;
  const int colh = h * HEADDIM_;
  const long vt_off = (long)(b * HIDDEN_ + colh) * SEQ_;   // row (b*2048 + h*64), stride SEQ_

  // Q fragments (16 rows x 64 d = two 16x32 A-frags), loaded once
  AFrag qf[2];
  {
    const int qrow = q_base + w * 16 + l15;
    const unsigned short* qp = Q + seq_off + (long)qrow * HIDDEN_ + colh;
#pragma unroll
    for (int f = 0; f < 2; ++f)
#pragma unroll
      for (int i = 0; i < 8; ++i) {
        const int k = ((i < 4) ? (2 * i) : (8 + 2 * i)) + 8 * lhi + 32 * f;
        qf[f].u[i] = *(const unsigned int*)(qp + k);
      }
  }

  float m_run[8], l_run[8];
#pragma unroll
  for (int r = 0; r < 8; ++r) { m_run[r] = -1e30f; l_run[r] = 0.0f; }
  CFrag o[4];
#pragma unroll
  for (int t = 0; t < 4; ++t)
#pragma unroll
    for (int r = 0; r < 8; ++r) o[t].f[r] = 0.0f;

  auto stage = [&](int jb, int bufi) {
    const int j = jb << 5;
#if USE_TDM
    if (w == 0) {
      tdm_load_tile_bf16(&Ks[bufi][0][0], Kk + seq_off + (long)j * HIDDEN_ + colh, 64u, 32u, (unsigned)HIDDEN_);
      tdm_load_tile_bf16(&Vs[bufi][0][0], Vt + vt_off + j,                          32u, 64u, (unsigned)SEQ_);
    }
#else
    { // K: 32 rows x 64 d, 128 threads x 16 elems
      const int r = tid >> 2, c = (tid & 3) * 16;
      const unsigned short* src = Kk + seq_off + (long)(j + r) * HIDDEN_ + colh + c;
      *(uint4*)&Ks[bufi][r][c]     = *(const uint4*)src;
      *(uint4*)&Ks[bufi][r][c + 8] = *(const uint4*)(src + 8);
    }
    { // V^T: 64 rows x 32 keys, 128 threads x 16 elems
      const int r = tid >> 1, c = (tid & 1) * 16;
      const unsigned short* src = Vt + vt_off + (long)r * SEQ_ + j + c;
      *(uint4*)&Vs[bufi][r][c]     = *(const uint4*)src;
      *(uint4*)&Vs[bufi][r][c + 8] = *(const uint4*)(src + 8);
    }
#endif
  };

  const int nblocks = (q_base + 64) >> 5;   // causal upper bound for this q tile
  stage(0, 0);
#if USE_TDM
  if (w == 0) __builtin_amdgcn_s_wait_tensorcnt(0);
#endif
  __syncthreads();

  for (int jb = 0; jb < nblocks; ++jb) {
    const int cur = jb & 1, nxt = cur ^ 1;
    const int j = jb << 5;
    if (jb + 1 < nblocks) stage(jb + 1, nxt);

    // scores S = Q * K^T : 16 q x 32 keys, accumulate over d
    CFrag s[2];
#pragma unroll
    for (int nt = 0; nt < 2; ++nt)
#pragma unroll
      for (int r = 0; r < 8; ++r) s[nt].f[r] = 0.0f;

#pragma unroll
    for (int f = 0; f < 2; ++f) {
      AFrag kb[2];
#pragma unroll
      for (int nt = 0; nt < 2; ++nt) {
        const int n = nt * 16 + l15;
#pragma unroll
        for (int i = 0; i < 8; ++i) {
          const int kd = 32 * f + ((i < 4) ? (2 * i) : (8 + 2 * i)) + 8 * lhi;
          kb[nt].u[i] = *(const unsigned int*)&Ks[cur][n][kd];   // B(kd,n) = K[j+n][kd]
        }
      }
      s[0].v = __builtin_amdgcn_wmma_f32_16x16x32_bf16(false, qf[f].v, false, kb[0].v, (short)0, s[0].v, false, false);
      s[1].v = __builtin_amdgcn_wmma_f32_16x16x32_bf16(false, qf[f].v, false, kb[1].v, (short)0, s[1].v, false, false);
    }

    // online softmax (rows live in 16-lane halves of each C vgpr)
    float alpha[8];
#pragma unroll
    for (int r = 0; r < 8; ++r) {
      const int qrow = q_base + w * 16 + r + 8 * lhi;
      const int key0 = j + l15;
      const int key1 = j + 16 + l15;
      float v0 = s[0].f[r] * 0.125f;
      float v1 = s[1].f[r] * 0.125f;
      if (key0 > qrow) v0 = -1e30f;
      if (key1 > qrow) v1 = -1e30f;
      const float mx   = rowmax16(fmaxf(v0, v1));
      const float mnew = fmaxf(m_run[r], mx);
      const float p0 = __expf(v0 - mnew);
      const float p1 = __expf(v1 - mnew);
      const float sum = rowsum16(p0 + p1);
      const float a = __expf(m_run[r] - mnew);
      l_run[r] = l_run[r] * a + sum;
      m_run[r] = mnew;
      alpha[r] = a;
      Ps[w][r + 8 * lhi][l15]      = f32_to_bf16(p0);   // C-layout -> row-major P
      Ps[w][r + 8 * lhi][16 + l15] = f32_to_bf16(p1);
    }
#pragma unroll
    for (int t = 0; t < 4; ++t)
#pragma unroll
      for (int r = 0; r < 8; ++r) o[t].f[r] *= alpha[r];

    // P as A-fragment (16x32), from this wave's private LDS buffer
    AFrag pf;
#pragma unroll
    for (int i = 0; i < 8; ++i) {
      const int k = ((i < 4) ? (2 * i) : (8 + 2 * i)) + 8 * lhi;
      pf.u[i] = *(const unsigned int*)&Ps[w][l15][k];
    }
    // O += P * V : four 16x16 d tiles
#pragma unroll
    for (int dt = 0; dt < 4; ++dt) {
      AFrag vf;
      const int dcol = dt * 16 + l15;
#pragma unroll
      for (int i = 0; i < 8; ++i) {
        const int kk = ((i < 4) ? (2 * i) : (8 + 2 * i)) + 8 * lhi;
        vf.u[i] = *(const unsigned int*)&Vs[cur][dcol][kk];    // B(kk,d) = V^T[d][j+kk]
      }
      o[dt].v = __builtin_amdgcn_wmma_f32_16x16x32_bf16(false, pf.v, false, vf.v, (short)0, o[dt].v, false, false);
    }

#if USE_TDM
    if (w == 0) __builtin_amdgcn_s_wait_tensorcnt(0);
#endif
    __syncthreads();
  }

  // normalize and store bf16 [B,S,H*D]
#pragma unroll
  for (int dt = 0; dt < 4; ++dt)
#pragma unroll
    for (int r = 0; r < 8; ++r) {
      const int qrow = q_base + w * 16 + r + 8 * lhi;
      const float val = o[dt].f[r] / l_run[r];
      O[seq_off + (long)qrow * HIDDEN_ + colh + dt * 16 + l15] = f32_to_bf16(val);
    }
}

// ---------------------------------------------------------------------------
extern "C" void kernel_launch(void* const* d_in, const int* in_sizes, int n_in,
                              void* d_out, int out_size, void* d_ws, size_t ws_size,
                              hipStream_t stream) {
  (void)in_sizes; (void)n_in; (void)out_size; (void)ws_size;

  const float* hidden = (const float*)d_in[0];
  const float* w_down = (const float*)d_in[1];
  const float* w_q_up = (const float*)d_in[2];
  const float* w_k_up = (const float*)d_in[3];
  const float* w_v_up = (const float*)d_in[4];
  const float* w_out  = (const float*)d_in[5];
  float* out = (float*)d_out;

  unsigned short* ws = (unsigned short*)d_ws;
  size_t off = 0;
  unsigned short* h_bf = ws + off; off += (size_t)ROWS_ * HIDDEN_;
  unsigned short* wdT  = ws + off; off += (size_t)HIDDEN_ * HIDDEN_;   // [2048][2048]
  unsigned short* wqT  = ws + off; off += (size_t)HIDDEN_ * QCOMP_;    // [2048][1024]
  unsigned short* wkT  = ws + off; off += (size_t)HIDDEN_ * KVCOMP_;   // [2048][512]
  unsigned short* wvT  = ws + off; off += (size_t)HIDDEN_ * KVCOMP_;
  unsigned short* woT  = ws + off; off += (size_t)HIDDEN_ * HIDDEN_;
  unsigned short* comp = ws + off; off += (size_t)ROWS_ * HIDDEN_;
  unsigned short* qm   = ws + off; off += (size_t)ROWS_ * HIDDEN_;
  unsigned short* km   = ws + off; off += (size_t)ROWS_ * HIDDEN_;
  unsigned short* vt   = ws + off; off += (size_t)ROWS_ * HIDDEN_;     // [b][h*64+d][s]
  unsigned short* am   = ws + off; off += (size_t)ROWS_ * HIDDEN_;

  // activations: plain f32->bf16; weights: transpose once ([K][N] -> [N][K])
  cvt_f32_to_bf16<<<(ROWS_ * HIDDEN_) / 1024, 256, 0, stream>>>(hidden, h_bf, ROWS_ * HIDDEN_);
  cvt_transpose_f32_to_bf16<<<dim3(HIDDEN_ / 32, HIDDEN_ / 32), 256, 0, stream>>>(w_down, wdT, HIDDEN_, HIDDEN_);
  cvt_transpose_f32_to_bf16<<<dim3(HIDDEN_ / 32, QCOMP_ / 32), 256, 0, stream>>>(w_q_up, wqT, QCOMP_, HIDDEN_);
  cvt_transpose_f32_to_bf16<<<dim3(HIDDEN_ / 32, KVCOMP_ / 32), 256, 0, stream>>>(w_k_up, wkT, KVCOMP_, HIDDEN_);
  cvt_transpose_f32_to_bf16<<<dim3(HIDDEN_ / 32, KVCOMP_ / 32), 256, 0, stream>>>(w_v_up, wvT, KVCOMP_, HIDDEN_);
  cvt_transpose_f32_to_bf16<<<dim3(HIDDEN_ / 32, HIDDEN_ / 32), 256, 0, stream>>>(w_out, woT, HIDDEN_, HIDDEN_);

  const dim3 gGrid(HIDDEN_ / 128, ROWS_ / 128);  // (16, 32)

  // comp = hidden @ w_down  (K = 2048)
  gemm_bf16_wmma<1><<<gGrid, 256, 0, stream>>>(h_bf, wdT, comp, HIDDEN_, HIDDEN_, HIDDEN_, HIDDEN_);
  // q = comp[:, :1024] @ w_q_up  (K = 1024)
  gemm_bf16_wmma<1><<<gGrid, 256, 0, stream>>>(comp, wqT, qm, QCOMP_, HIDDEN_, QCOMP_, HIDDEN_);
  // k = comp[:, 1024:1536] @ w_k_up  (K = 512)
  gemm_bf16_wmma<1><<<gGrid, 256, 0, stream>>>(comp + QCOMP_, wkT, km, KVCOMP_, HIDDEN_, KVCOMP_, HIDDEN_);
  // v (output pre-transposed per batch: vt[b][h*64+d][s])  (K = 512)
  gemm_bf16_wmma<2><<<gGrid, 256, 0, stream>>>(comp + QCOMP_ + KVCOMP_, wvT, vt, KVCOMP_, HIDDEN_, KVCOMP_, HIDDEN_);

  // causal flash attention per (b, head)
  mla_flash_attn<<<dim3(SEQ_ / 64, BATCH_ * NHEADS_), 128, 0, stream>>>(qm, km, vt, am);

  // out = attn @ w_out  (K = 2048), f32 output
  gemm_bf16_wmma<0><<<gGrid, 256, 0, stream>>>(am, woT, (void*)out, HIDDEN_, HIDDEN_, HIDDEN_, HIDDEN_);
}
